// CausalSE_8890582302883
// MI455X (gfx1250) — matrix operand order repaired and checked by reading, not compiled, not run on hardware
//
#include <hip/hip_runtime.h>
#include <hip/hip_bf16.h>

// ---- fixed problem shape (from reference setup_inputs) ----
#define B_  8
#define C_  512
#define L_  8192
#define N_  512    // L / CHUNK(16)
#define H_  64     // C / 8

typedef _Float16 h8   __attribute__((ext_vector_type(8)));
typedef _Float16 h16  __attribute__((ext_vector_type(16)));
typedef float    f8v  __attribute__((ext_vector_type(8)));

// ---------------- K0: convert SE weights to f16 ----------------
__global__ void wcvt_kernel(const float* __restrict__ w1, const float* __restrict__ w2,
                            _Float16* __restrict__ w1h, _Float16* __restrict__ w2h) {
    int i = blockIdx.x * 256 + threadIdx.x;     // 0 .. 64*512-1 (both weights same size)
    w1h[i] = (_Float16)w1[i];
    w2h[i] = (_Float16)w2[i];
}

// ---------------- K1: chunked mean pool  x[B,C,L] -> xc[B,N,C] ----------------
__global__ void pool_kernel(const float* __restrict__ x, float* __restrict__ xc) {
    int idx = blockIdx.x * 256 + threadIdx.x;   // idx = bc*N + n  (lanes contiguous in n)
    int bc  = idx >> 9;
    int n   = idx & (N_ - 1);
    const float4* xp = (const float4*)(x + (size_t)bc * L_ + (size_t)n * 16);
    float4 a = xp[0], b = xp[1], c = xp[2], d = xp[3];
    float s = (a.x + a.y + a.z + a.w) + (b.x + b.y + b.z + b.w)
            + (c.x + c.y + c.z + c.w) + (d.x + d.y + d.z + d.w);
    int bb = bc >> 9, cc = bc & (C_ - 1);
    xc[((size_t)bb * N_ + n) * C_ + cc] = s * (1.0f / 16.0f);
}

// ---------------- K2: causal EMA over chunks, emit f16 eT[B,N,C] ----------------
__global__ void ema_kernel(const float* __restrict__ xc, const float* __restrict__ gamma,
                           _Float16* __restrict__ eT) {
    int t = blockIdx.x * 256 + threadIdx.x;     // t = b*C + c  (lanes contiguous in c)
    int b = t >> 9, c = t & (C_ - 1);
    float g = gamma[c];
    float om = 1.0f - g;
    float y = 0.0f;
    size_t base = (size_t)b * N_ * C_ + c;
    for (int n = 0; n < N_; ++n) {
        y = g * y + om * xc[base + (size_t)n * C_];
        eT[base + (size_t)n * C_] = (_Float16)y;
    }
}

// ---------------- K3: fused SE bottleneck via WMMA + async-to-LDS staging ----------------
// grid = (N/64, B), block = 256 (8 waves of 32).
// At entry: kick off async copy of w2h (64 KB) into LDS (ASYNCcnt-tracked),
//           overlapped with stage-1 WMMA that reads only w1h/eT from global.
// Stage 1: h[64, 64cols] = relu(w1 @ e + b1)      (K = 512, 16 wmma k-steps)
// Stage 2: gate = sigmoid(w2 @ h + b2)            (K = 64,  2 wmma k-steps, A+B from LDS)
__global__ __launch_bounds__(256) void se_kernel(
    const _Float16* __restrict__ eT,  const _Float16* __restrict__ w1h,
    const _Float16* __restrict__ w2h, const float* __restrict__ b1,
    const float* __restrict__ b2,     float* __restrict__ gate) {

    // h transposed: hT[n_local][h_channel]; stride 80 halves = 160 B (multiple of 32 B)
    __shared__ __align__(32) _Float16 hT[64][80];
    // staged w2: [512 rows][64 + 8 pad halves]; stride 144 B keeps b128 frags 16B-aligned
    __shared__ __align__(32) _Float16 w2L[512][72];

    const int b     = blockIdx.y;
    const int ncol0 = blockIdx.x * 64;
    const int tid   = threadIdx.x;
    const int wid   = tid >> 5;
    const int lane  = tid & 31;
    const int hl    = lane >> 4;     // lane half (selects K sub-range in A/B fragments)
    const int ln    = lane & 15;     // row (A) / column (B,C,D) index within tile

    // ---- issue async LDS staging of w2h: 4096 x 16B chunks, 16 per thread ----
    {
        // generic cast of an LDS pointer carries the workgroup-relative LDS byte
        // offset in the low 32 bits (SHARED aperture lives in addr[63:32]).
        unsigned ldsBase = (unsigned)(uintptr_t)(void*)&w2L[0][0];
#pragma unroll
        for (int i = 0; i < 16; ++i) {
            int q   = tid + 256 * i;          // chunk id: row = q/8, 16B-chunk-in-row = q%8
            int row = q >> 3, cc = q & 7;
            unsigned    dst = ldsBase + (unsigned)(row * 144 + cc * 16);
            const void* src = (const char*)w2h + (size_t)row * 128 + (size_t)cc * 16;
            asm volatile("global_load_async_to_lds_b128 %0, %1, off"
                         :: "v"(dst), "v"(src) : "memory");
        }
    }

    const _Float16* eB = eT + (size_t)b * N_ * C_;

    // ---- Stage 1: 16 output tiles (4 mt x 4 nt), 2 tiles per wave ----
    for (int t = 2 * wid; t < 2 * wid + 2; ++t) {
        const int mt = t >> 2, nt = t & 3;
        f8v acc = {};
        const int arow = mt * 16 + ln;             // row of w1 (h-channel)
        const int ncol = ncol0 + nt * 16 + ln;     // global chunk-column
        for (int kk = 0; kk < 16; ++kk) {          // K = 512 = 16 * 32
            union { h16 v; h8 h2[2]; } A;
            const _Float16* ap = w1h + (size_t)arow * C_ + kk * 32 + hl * 16;
            A.h2[0] = *(const h8*)ap;              // K = base + 0..7
            A.h2[1] = *(const h8*)(ap + 8);        // K = base + 8..15
            h16 Bf = *(const h16*)(eB + (size_t)ncol * C_ + kk * 32 + hl * 16);
            acc = __builtin_amdgcn_wmma_f32_16x16x32_f16(
                      false, A.v, false, Bf, (short)0, acc, false, false);
        }
        // bias + relu, transpose into LDS (8 consecutive h-channels -> one b128 store)
        const int hbase = mt * 16 + 8 * hl;        // C/D row = r + 8*(lane>=16)
        const int nl    = nt * 16 + ln;
        union { h8 v; _Float16 e[8]; } hv;
#pragma unroll
        for (int r = 0; r < 8; ++r) {
            float v = acc[r] + b1[hbase + r];
            hv.e[r] = (_Float16)(v > 0.0f ? v : 0.0f);
        }
        *(h8*)&hT[nl][hbase] = hv.v;
    }

    // async staging must be complete (ASYNCcnt) before anyone reads w2L
    asm volatile("s_wait_asynccnt 0x0" ::: "memory");
    __syncthreads();

    // ---- Stage 2: 128 output tiles (32 mt x 4 nt), 16 tiles per wave; A+B from LDS ----
    for (int j = 0; j < 16; ++j) {
        const int T  = wid * 16 + j;
        const int mt = T >> 2, nt = T & 3;
        f8v acc = {};
        const int arow = mt * 16 + ln;             // row of w2 (output channel)
        const int nl   = nt * 16 + ln;
#pragma unroll
        for (int kk = 0; kk < 2; ++kk) {           // K = 64 = 2 * 32
            union { h16 v; h8 h2[2]; } A;
            const _Float16* ap = &w2L[arow][kk * 32 + hl * 16];
            A.h2[0] = *(const h8*)ap;              // ds_read_b128
            A.h2[1] = *(const h8*)(ap + 8);
            h16 Bf = *(const h16*)&hT[nl][kk * 32 + hl * 16];
            acc = __builtin_amdgcn_wmma_f32_16x16x32_f16(
                      false, A.v, false, Bf, (short)0, acc, false, false);
        }
        const int ch0 = mt * 16 + 8 * hl;
        const int ng  = ncol0 + nt * 16 + ln;
#pragma unroll
        for (int r = 0; r < 8; ++r) {
            float o = acc[r] + b2[ch0 + r];
            float s = 1.0f / (1.0f + __expf(-o));
            gate[((size_t)b * C_ + (ch0 + r)) * N_ + ng] = s;
        }
    }
}

// ---------------- K4: out = repeat(gate,16) * x  (bandwidth-bound tail) ----------------
__global__ void apply_kernel(const float* __restrict__ x, const float* __restrict__ gate,
                             float* __restrict__ out) {
    size_t i4 = (size_t)blockIdx.x * 256 + threadIdx.x;   // float4 index
    size_t e0 = i4 * 4;
    size_t bc = e0 >> 13;                                 // / L_
    int    l  = (int)(e0 & (L_ - 1));
    float  g  = gate[bc * N_ + (l >> 4)];                 // 4 elems share one chunk
    float4 xv = ((const float4*)x)[i4];
    float4 ov;  ov.x = xv.x * g;  ov.y = xv.y * g;  ov.z = xv.z * g;  ov.w = xv.w * g;
    ((float4*)out)[i4] = ov;
}

extern "C" void kernel_launch(void* const* d_in, const int* in_sizes, int n_in,
                              void* d_out, int out_size, void* d_ws, size_t ws_size,
                              hipStream_t stream) {
    const float* x     = (const float*)d_in[0];   // [B,C,L]
    const float* gamma = (const float*)d_in[1];   // [C]
    const float* w1    = (const float*)d_in[2];   // [H,C]
    const float* b1    = (const float*)d_in[3];   // [H]
    const float* w2    = (const float*)d_in[4];   // [C,H]
    const float* b2    = (const float*)d_in[5];   // [C]
    float* out = (float*)d_out;

    // workspace layout (bytes)
    char* ws = (char*)d_ws;
    float*    xc   = (float*)   (ws);                         //  8 MB  [B,N,C] f32
    _Float16* eTh  = (_Float16*)(ws + 8388608);               //  4 MB  [B,N,C] f16
    _Float16* w1h  = (_Float16*)(ws + 12582912);              // 64 KB
    _Float16* w2h  = (_Float16*)(ws + 12648448);              // 64 KB
    float*    gate = (float*)   (ws + 12713984);              //  8 MB  [B,C,N] f32

    wcvt_kernel <<< (H_ * C_) / 256, 256, 0, stream >>> (w1, w2, w1h, w2h);
    pool_kernel <<< (B_ * C_ * N_) / 256, 256, 0, stream >>> (x, xc);
    ema_kernel  <<< (B_ * C_) / 256, 256, 0, stream >>> (xc, gamma, eTh);
    se_kernel   <<< dim3(N_ / 64, B_), 256, 0, stream >>> (eTh, w1h, w2h, b1, b2, gate);
    apply_kernel<<< (B_ * C_ * (L_ / 4)) / 256, 256, 0, stream >>> (x, gate, out);
}
